// CrossViewSwapAttention_44006234915575
// MI455X (gfx1250) — matrix operand order repaired and checked by reading, not compiled, not run on hardware
//
#include <hip/hip_runtime.h>

typedef __bf16 bf16;
typedef __attribute__((ext_vector_type(16))) __bf16 v16bf;
typedef __attribute__((ext_vector_type(2)))  __bf16 bf16x2;
typedef __attribute__((ext_vector_type(8)))  float  v8f;

#define WMMA_BF16(a, b, c) \
  __builtin_amdgcn_wmma_f32_16x16x32_bf16(false, (a), false, (b), (short)0, (c), false, false)

// In-wave LDS ordering: DS ops complete in order per wave; stop compiler
// reordering and wait out outstanding LDS before cross-lane reuse of a buffer.
static __device__ __forceinline__ void wave_lds_fence() {
  asm volatile("s_wait_dscnt 0" ::: "memory");
}

// A fragment (16x32 bf16, row-major source, leading dim `ld` elements).
// Layout used here (contiguous k, same as B): lanes 0-15 -> M=lane, K=0..15;
//             lanes 16-31 -> M=lane-16, K=16..31.
static __device__ __forceinline__ v16bf load_a16x32(const bf16* base, int ld, int lane) {
  const int half = (lane >> 4) & 1;
  const bf16* row = base + (lane & 15) * ld;
  const int kA = half * 16;
  const int kB = half * 16 + 8;
  v16bf a;
#pragma unroll
  for (int j = 0; j < 4; ++j) {
    bf16x2 p0 = *(const bf16x2*)(row + kA + 2 * j);
    bf16x2 p1 = *(const bf16x2*)(row + kB + 2 * j);
    a[2 * j] = p0.x;     a[2 * j + 1] = p0.y;
    a[8 + 2 * j] = p1.x; a[8 + 2 * j + 1] = p1.y;
  }
  return a;
}

// B fragment (32x16 bf16) from N-major storage: element (k,n) at base[n*ld + k].
// Layout used here (contiguous k, same as A): lanes 0-15 -> N=lane, K=0..15 (2 per VGPR); lanes 16-31 -> K=16..31.
static __device__ __forceinline__ v16bf load_b32x16(const bf16* base, int ld, int lane) {
  const bf16* row = base + (lane & 15) * ld;
  const int kb = ((lane >> 4) & 1) * 16;
  v16bf b;
#pragma unroll
  for (int j = 0; j < 8; ++j) {
    bf16x2 p = *(const bf16x2*)(row + kb + 2 * j);
    b[2 * j] = p.x; b[2 * j + 1] = p.y;
  }
  return b;
}

// LayerNorm one 128-wide row across the 32 lanes of a wave (4 cols/lane),
// write bf16 row to LDS.
static __device__ __forceinline__ void ln_row(const float* __restrict__ src,
                                              const float* __restrict__ g,
                                              const float* __restrict__ bta,
                                              bf16* dst, int lane) {
  float4 x = ((const float4*)src)[lane];
  float s = x.x + x.y + x.z + x.w;
  float q = x.x * x.x + x.y * x.y + x.z * x.z + x.w * x.w;
#pragma unroll
  for (int o = 16; o >= 1; o >>= 1) {
    s += __shfl_xor(s, o, 32);
    q += __shfl_xor(q, o, 32);
  }
  const float mean = s * (1.0f / 128.0f);
  const float var  = q * (1.0f / 128.0f) - mean * mean;
  const float inv  = rsqrtf(var + 1e-5f);
  float4 gv = ((const float4*)g)[lane];
  float4 bv = ((const float4*)bta)[lane];
  const int col = lane * 4;
  dst[col + 0] = (bf16)((x.x - mean) * inv * gv.x + bv.x);
  dst[col + 1] = (bf16)((x.y - mean) * inv * gv.y + bv.y);
  dst[col + 2] = (bf16)((x.z - mean) * inv * gv.z + bv.z);
  dst[col + 3] = (bf16)((x.w - mean) * inv * gv.w + bv.w);
}

__global__ __launch_bounds__(256, 1)
void cvswap_attn_kernel(const float* __restrict__ q, const float* __restrict__ k,
                        const float* __restrict__ v, const float* __restrict__ skip,
                        const float* __restrict__ head_gate,
                        const float* __restrict__ lnqg, const float* __restrict__ lnqb,
                        const float* __restrict__ lnkg, const float* __restrict__ lnkb,
                        const float* __restrict__ lnvg, const float* __restrict__ lnvb,
                        const float* __restrict__ wq, const float* __restrict__ bq,
                        const float* __restrict__ wk, const float* __restrict__ bk,
                        const float* __restrict__ wv, const float* __restrict__ bv,
                        const float* __restrict__ wp, const float* __restrict__ bp,
                        float* __restrict__ out) {
  constexpr int D = 128;   // model dim == inner dim
  constexpr int NV = 6;    // views
  constexpr int L = 100;   // windows (X*Y == x*y)
  constexpr int POS = 100; // positions per window (W1*W2)

  __shared__ alignas(16) bf16 sWa[D * D];        // wk^T then wq^T   (32 KB)
  __shared__ alignas(16) bf16 sWb[D * D];        // wv^T then wp^T   (32 KB)
  __shared__ alignas(16) bf16 sKh[96 * D];       // kh [token][dim]  (24 KB)
  __shared__ alignas(16) bf16 sVt[D * 96];       // vh^T [dim][token](24 KB)
  __shared__ alignas(16) bf16 sBuf[8][16 * D];   // per-wave staging (32 KB)

  const int tid = threadIdx.x;
  const int wave = tid >> 5, lane = tid & 31;
  const int half = (lane >> 4) & 1, lane15 = lane & 15;
  const int bb = blockIdx.x / L, l = blockIdx.x % L;

  // ---- stage wk^T / wv^T as bf16 ----
  for (int i = tid; i < D * D; i += 256) {
    const int kk = i >> 7, nn = i & 127;
    sWa[nn * D + kk] = (bf16)wk[i];
    sWb[nn * D + kk] = (bf16)wv[i];
  }
  __syncthreads();

  // ---- Phase A: kh / vh (96 kv tokens, 6 tiles, waves 0..5) ----
  if (wave < 6) {
    bf16* buf = sBuf[wave];
    const int kt = wave;
    // K: LN -> project -> sKh (token-major)
    for (int r = 0; r < 16; ++r) {
      const int t = kt * 16 + r, nk = t >> 4, t16 = t & 15;
      ln_row(k + ((size_t)(((bb * NV + nk) * L + l) * 16 + t16)) * D, lnkg, lnkb,
             buf + r * D, lane);
    }
    wave_lds_fence();
    v16bf aF[4];
#pragma unroll
    for (int kk = 0; kk < 4; ++kk) aF[kk] = load_a16x32(buf + kk * 32, D, lane);
#pragma unroll
    for (int nt = 0; nt < 8; ++nt) {
      v8f acc = {};
#pragma unroll
      for (int kk = 0; kk < 4; ++kk)
        acc = WMMA_BF16(aF[kk], load_b32x16(sWa + (nt * 16) * D + kk * 32, D, lane), acc);
      const float bias = bk[nt * 16 + lane15];
#pragma unroll
      for (int r = 0; r < 8; ++r)
        sKh[(kt * 16 + r + 8 * half) * D + nt * 16 + lane15] = (bf16)(acc[r] + bias);
    }
    wave_lds_fence();
    // V: LN -> project -> sVt (dim-major / transposed)
    for (int r = 0; r < 16; ++r) {
      const int t = kt * 16 + r, nk = t >> 4, t16 = t & 15;
      ln_row(v + ((size_t)(((bb * NV + nk) * L + l) * 16 + t16)) * D, lnvg, lnvb,
             buf + r * D, lane);
    }
    wave_lds_fence();
#pragma unroll
    for (int kk = 0; kk < 4; ++kk) aF[kk] = load_a16x32(buf + kk * 32, D, lane);
#pragma unroll
    for (int nt = 0; nt < 8; ++nt) {
      v8f acc = {};
#pragma unroll
      for (int kk = 0; kk < 4; ++kk)
        acc = WMMA_BF16(aF[kk], load_b32x16(sWb + (nt * 16) * D + kk * 32, D, lane), acc);
      const float bias = bv[nt * 16 + lane15];
#pragma unroll
      for (int r = 0; r < 8; ++r)
        sVt[(nt * 16 + lane15) * 96 + kt * 16 + r + 8 * half] = (bf16)(acc[r] + bias);
    }
  }
  __syncthreads();

  // ---- restage wq^T / wp^T ----
  for (int i = tid; i < D * D; i += 256) {
    const int kk = i >> 7, nn = i & 127;
    sWa[nn * D + kk] = (bf16)wq[i];
    sWb[nn * D + kk] = (bf16)wp[i];
  }
  __syncthreads();

  // ---- Phase B: each wave owns one 16-position tile; loop views in regs ----
  if (wave >= 7) return;
  bf16* buf = sBuf[wave];
  const int pt = wave;
  const float scale = 0.17677669529663687f;  // 1/sqrt(32)

  v8f zacc[8];
  {
    v8f z = {};
#pragma unroll
    for (int i = 0; i < 8; ++i) zacc[i] = z;
  }

#pragma unroll 1
  for (int n = 0; n < NV; ++n) {
    // LN(q rows) -> buf
    for (int r = 0; r < 16; ++r) {
      const int pos = pt * 16 + r;
      if (pos < POS) {
        ln_row(q + ((size_t)(((bb * NV + n) * L + l) * POS + pos)) * D, lnqg, lnqb,
               buf + r * D, lane);
      } else {
        const int col = lane * 4;
        const bf16 z0 = (bf16)0.0f;
        buf[r * D + col] = z0; buf[r * D + col + 1] = z0;
        buf[r * D + col + 2] = z0; buf[r * D + col + 3] = z0;
      }
    }
    wave_lds_fence();
    // qh = LN(q) @ wq + bq  -> buf (bf16)
    v16bf aq[4];
#pragma unroll
    for (int kk = 0; kk < 4; ++kk) aq[kk] = load_a16x32(buf + kk * 32, D, lane);
#pragma unroll
    for (int nt = 0; nt < 8; ++nt) {
      v8f acc = {};
#pragma unroll
      for (int kk = 0; kk < 4; ++kk)
        acc = WMMA_BF16(aq[kk], load_b32x16(sWa + (nt * 16) * D + kk * 32, D, lane), acc);
      const float bias = bq[nt * 16 + lane15];
#pragma unroll
      for (int r = 0; r < 8; ++r)
        buf[(r + 8 * half) * D + nt * 16 + lane15] = (bf16)(acc[r] + bias);
    }
    wave_lds_fence();
    v16bf ah[4];
#pragma unroll
    for (int m = 0; m < 4; ++m) ah[m] = load_a16x32(buf + m * 32, D, lane);

    v8f aacc[8];
    {
      v8f z = {};
#pragma unroll
      for (int i = 0; i < 8; ++i) aacc[i] = z;
    }

#pragma unroll
    for (int m = 0; m < 4; ++m) {
      // scores S = qh_m @ kh_m^T  (16 x 96)
      v8f sc[6];
#pragma unroll
      for (int t = 0; t < 6; ++t) {
        v8f zero = {};
        sc[t] = WMMA_BF16(ah[m], load_b32x16(sKh + (t * 16) * D + m * 32, D, lane), zero);
      }
      const float fac = scale * head_gate[m];
      // softmax along the 96 keys (rows live in 16-lane halves)
#pragma unroll
      for (int r = 0; r < 8; ++r) {
        float mx = -3.0e38f;
#pragma unroll
        for (int t = 0; t < 6; ++t) { float sv = sc[t][r] * fac; sc[t][r] = sv; mx = fmaxf(mx, sv); }
#pragma unroll
        for (int o = 1; o <= 8; o <<= 1) mx = fmaxf(mx, __shfl_xor(mx, o, 32));
        float ssum = 0.0f;
#pragma unroll
        for (int t = 0; t < 6; ++t) { float e = __expf(sc[t][r] - mx); sc[t][r] = e; ssum += e; }
#pragma unroll
        for (int o = 1; o <= 8; o <<= 1) ssum += __shfl_xor(ssum, o, 32);
        const float inv = 1.0f / ssum;
#pragma unroll
        for (int t = 0; t < 6; ++t) sc[t][r] *= inv;
      }
      // P -> buf (16 x 96, stride 96), then O_m = P @ vh_m
#pragma unroll
      for (int t = 0; t < 6; ++t)
#pragma unroll
        for (int r = 0; r < 8; ++r)
          buf[(r + 8 * half) * 96 + t * 16 + lane15] = (bf16)sc[t][r];
      wave_lds_fence();
#pragma unroll
      for (int kb = 0; kb < 3; ++kb) {
        v16bf aP = load_a16x32(buf + kb * 32, 96, lane);
#pragma unroll
        for (int nt2 = 0; nt2 < 2; ++nt2) {
          v16bf bV = load_b32x16(sVt + (m * 32 + nt2 * 16) * 96 + kb * 32, 96, lane);
          aacc[2 * m + nt2] = WMMA_BF16(aP, bV, aacc[2 * m + nt2]);
        }
      }
      wave_lds_fence();
    }
    // a -> buf, then zacc += a @ wp  (accumulates over views too)
#pragma unroll
    for (int nt = 0; nt < 8; ++nt)
#pragma unroll
      for (int r = 0; r < 8; ++r)
        buf[(r + 8 * half) * D + nt * 16 + lane15] = (bf16)aacc[nt][r];
    wave_lds_fence();
    v16bf aA[4];
#pragma unroll
    for (int kk = 0; kk < 4; ++kk) aA[kk] = load_a16x32(buf + kk * 32, D, lane);
#pragma unroll
    for (int nt = 0; nt < 8; ++nt) {
      v8f c = zacc[nt];
#pragma unroll
      for (int kk = 0; kk < 4; ++kk)
        c = WMMA_BF16(aA[kk], load_b32x16(sWb + (nt * 16) * D + kk * 32, D, lane), c);
      zacc[nt] = c;
    }
    wave_lds_fence();
  }

  // out = mean_n(z) + bp + skip
#pragma unroll
  for (int nt = 0; nt < 8; ++nt) {
    const int col = nt * 16 + lane15;
    const float bpv = bp[col];
#pragma unroll
    for (int r = 0; r < 8; ++r) {
      const int pos = pt * 16 + r + 8 * half;
      if (pos < POS) {
        const size_t idx = ((size_t)(bb * L + l) * POS + pos) * D + col;
        out[idx] = zacc[nt][r] * (1.0f / 6.0f) + bpv + skip[idx];
      }
    }
  }
}

extern "C" void kernel_launch(void* const* d_in, const int* in_sizes, int n_in,
                              void* d_out, int out_size, void* d_ws, size_t ws_size,
                              hipStream_t stream) {
  (void)in_sizes; (void)n_in; (void)out_size; (void)d_ws; (void)ws_size;
  const float* q    = (const float*)d_in[0];
  const float* k    = (const float*)d_in[1];
  const float* v    = (const float*)d_in[2];
  const float* skip = (const float*)d_in[3];
  const float* hg   = (const float*)d_in[4];
  const float* lnqg = (const float*)d_in[5];
  const float* lnqb = (const float*)d_in[6];
  const float* lnkg = (const float*)d_in[7];
  const float* lnkb = (const float*)d_in[8];
  const float* lnvg = (const float*)d_in[9];
  const float* lnvb = (const float*)d_in[10];
  const float* wq   = (const float*)d_in[11];
  const float* bq   = (const float*)d_in[12];
  const float* wk   = (const float*)d_in[13];
  const float* bk   = (const float*)d_in[14];
  const float* wv   = (const float*)d_in[15];
  const float* bv   = (const float*)d_in[16];
  const float* wp   = (const float*)d_in[17];
  const float* bp   = (const float*)d_in[18];
  float* out = (float*)d_out;

  dim3 grid(4 * 100);   // (batch, window)
  dim3 block(256);      // 8 wave32
  hipLaunchKernelGGL(cvswap_attn_kernel, grid, block, 0, stream,
                     q, k, v, skip, hg, lnqg, lnqb, lnkg, lnkb, lnvg, lnvb,
                     wq, bq, wk, bk, wv, bv, wp, bp, out);
}
